// EncoderAttentionNodes_46273977647528
// MI455X (gfx1250) — matrix operand order repaired and checked by reading, not compiled, not run on hardware
//
#include <hip/hip_runtime.h>
#include <hip/hip_bf16.h>

typedef __attribute__((ext_vector_type(16))) _Float16 v16h;
typedef __attribute__((ext_vector_type(8)))  _Float16 v8h;
typedef __attribute__((ext_vector_type(8)))  float    v8f;

#define HW   128
#define TILE 32
#define XD   40   // x tile extent (38 + 2 halo for conv1, origin at -4 due to pad 4)
#define H1D  38
#define H2D  36
#define H3D  34

// ---- LDS layout (bytes) ----
#define ASWZ_OFF   0
#define ASWZ_BYTES (36 * 512 * 2)              // 36864: up to 36 A-fragment blocks (32 lanes x 16 f16)
#define H1_OFF     (ASWZ_OFF + ASWZ_BYTES)     // h1: 38*38 pos x 64ch f16 ; later reused for h3 (34*34 x 32ch)
#define H1_BYTES   (H1D * H1D * 64 * 2)        // 184832
#define H2_OFF     (H1_OFF + H1_BYTES)         // h2: 36*36 pos x 32ch f16 ; x-tile overlaps its start
#define H2_BYTES   (H2D * H2D * 32 * 2)        // 82944
#define CST_OFF    (H2_OFF + H2_BYTES)
// const region (float index):
//  [0..575]     w1 (64x9) fp32
//  [576..639]   scale1   [640..703] shift1
//  [704..735]   scale2   [736..767] shift2
//  [768..799]   scale3   [800..831] shift3
//  [832..1119]  w4 (32x9) fp32
//  [1120]       b4
//  [1121..1136] cellMax  [1137..1152] cellSum
//  [1153..1168] cellSx   [1169..1184] cellSy
#define CST_FLOATS 1192
#define SMEM_BYTES (CST_OFF + CST_FLOATS * 4)  // ~309 KB of the 320 KB WGP LDS

// d_out layout (floats, tuple concat: nodes, raw, valid, nodes_per_image)
#define OUT_NODES 0
#define OUT_RAW   (128 * 256 * 2)                       // 65536
#define OUT_VALID (OUT_RAW + 128 * HW * HW)             // 2162688
#define OUT_NPI   (OUT_VALID + 128 * 256)               // 2195456

static __device__ __forceinline__ v8f wmma_f16(v16h a, v16h b, v8f c) {
    return __builtin_amdgcn_wmma_f32_16x16x32_f16(false, a, false, b, (short)0, c,
                                                  false, false);
}

// A-fragment K index for (lane, slot): contiguous order k = 16*(l/16) + s (matches B loads).
static __device__ __forceinline__ int a_k_of(int l, int s) {
    return (l < 16) ? s : s + 16;
}

// ---- CDNA5 async global->LDS (ASYNCcnt-tracked, no VGPR round trip) ----
typedef __attribute__((address_space(1))) int gas_int;
typedef __attribute__((address_space(3))) int las_int;

static __device__ __forceinline__ void async_load_f32(float* ldsDst, const float* gsrc) {
#if __has_builtin(__builtin_amdgcn_global_load_async_to_lds_b32)
    __builtin_amdgcn_global_load_async_to_lds_b32(
        (gas_int*)gsrc, (las_int*)ldsDst, 0, 0);
#else
    // LDS generic pointers: addr[31:0] is the LDS byte address (aperture truncation)
    unsigned ldsAddr = (unsigned)(unsigned long long)ldsDst;
    asm volatile("global_load_async_to_lds_b32 %0, %1, off"
                 :: "v"(ldsAddr), "v"(gsrc) : "memory");
#endif
}

static __device__ __forceinline__ void wait_asynccnt0() {
#if __has_builtin(__builtin_amdgcn_s_wait_asynccnt)
    __builtin_amdgcn_s_wait_asynccnt(0);
#else
    asm volatile("s_wait_asynccnt 0" ::: "memory");
#endif
}

__global__ __launch_bounds__(256)
void fused_conv_tile(const float* __restrict__ x,
                     const float* __restrict__ w1, const float* __restrict__ b1,
                     const float* __restrict__ g1, const float* __restrict__ bt1,
                     const float* __restrict__ rm1, const float* __restrict__ rv1,
                     const float* __restrict__ w2, const float* __restrict__ b2,
                     const float* __restrict__ g2, const float* __restrict__ bt2,
                     const float* __restrict__ rm2, const float* __restrict__ rv2,
                     const float* __restrict__ w3, const float* __restrict__ b3,
                     const float* __restrict__ g3, const float* __restrict__ bt3,
                     const float* __restrict__ rm3, const float* __restrict__ rv3,
                     const float* __restrict__ w4, const float* __restrict__ b4,
                     float* __restrict__ rawOut, float* __restrict__ stats) {
    extern __shared__ __align__(32) char smem[];
    _Float16* aswz = (_Float16*)(smem + ASWZ_OFF);
    _Float16* h1s  = (_Float16*)(smem + H1_OFF);
    _Float16* h3s  = (_Float16*)(smem + H1_OFF);   // reuse after h1 dead
    _Float16* h2s  = (_Float16*)(smem + H2_OFF);
    float*    xt   = (float*)(smem + H2_OFF);      // overlap: x dead before h2 written
    float*    cst  = (float*)(smem + CST_OFF);

    const int tid  = threadIdx.x;
    const int wid  = tid >> 5;
    const int lane = tid & 31;
    const int ln   = lane & 15;
    const int lh   = lane >> 4;

    const int b    = blockIdx.x >> 4;
    const int tIdx = blockIdx.x & 15;
    const int ty   = tIdx >> 2, tx = tIdx & 3;
    const int gy0  = ty * TILE, gx0 = tx * TILE;

    // Pull weight tensors toward cache early (global_prefetch_b8).
    __builtin_prefetch(w2 + tid, 0, 0);
    __builtin_prefetch(w3 + tid, 0, 0);

    // ---- Phase 0a: x tile via async global->LDS; OOB entries zero-filled via ds ----
    const float* xb = x + (size_t)b * HW * HW;
    for (int i = tid; i < XD * XD; i += 256) {
        int r = (i / XD) - 4 + gy0;
        int c = (i % XD) - 4 + gx0;
        if (r >= 0 && r < HW && c >= 0 && c < HW) {
            async_load_f32(&xt[i], xb + r * HW + c);
        } else {
            xt[i] = 0.0f;
        }
    }

    // ---- Phase 0b: constants, BN folds, conv2 A-fragment swizzle ----
    for (int i = tid; i < 64 * 9; i += 256) cst[i] = w1[i];
    for (int i = tid; i < 32 * 9; i += 256) cst[832 + i] = w4[i];
    if (tid == 0) cst[1120] = b4[0];
    if (tid < 64) {
        float inv = g1[tid] * rsqrtf(rv1[tid] + 1e-5f);
        cst[576 + tid] = inv;
        cst[640 + tid] = b1[tid] * inv + bt1[tid] - rm1[tid] * inv;
    } else if (tid < 96) {
        int c = tid - 64;
        float inv = g2[c] * rsqrtf(rv2[c] + 1e-5f);
        cst[704 + c] = inv;
        cst[736 + c] = b2[c] * inv + bt2[c] - rm2[c] * inv;
    } else if (tid < 128) {
        int c = tid - 96;
        float inv = g3[c] * rsqrtf(rv3[c] + 1e-5f);
        cst[768 + c] = inv;
        cst[800 + c] = b3[c] * inv + bt3[c] - rm3[c] * inv;
    } else if (tid < 144) {
        int c = tid - 128;
        cst[1121 + c] = 0.0f;   // sigmoid > 0, so uint-max on 0.0 works
        cst[1137 + c] = 0.0f;
        cst[1153 + c] = 0.0f;
        cst[1169 + c] = 0.0f;
    }

    // conv2 A fragments: block = (dd*2+cc)*2+mt, K chunk = channels cc*32..cc*32+31
    for (int i = tid; i < 36 * 512; i += 256) {
        int blk = i >> 9, l = (i >> 4) & 31, s = i & 15;
        int mt = blk & 1, cc = (blk >> 1) & 1, dd = blk >> 2;
        int K  = a_k_of(l, s);
        int ch = cc * 32 + K;
        int m  = mt * 16 + (l & 15);
        aswz[i] = (_Float16)w2[(m * 64 + ch) * 9 + dd];
    }
    wait_asynccnt0();   // x tile async fills complete before the barrier
    __syncthreads();

    // ---- Phase 1: conv1 (1->64, K=9) fp32 VALU + BN + ReLU -> h1 f16 channel-last ----
    for (int i = tid; i < H1D * H1D * 64; i += 256) {
        int pos = i >> 6, ch = i & 63;
        int py = pos / H1D, px = pos % H1D;
        float acc = 0.0f;
#pragma unroll
        for (int dd = 0; dd < 9; ++dd) {
            acc += xt[(py + dd / 3) * XD + (px + dd % 3)] * cst[ch * 9 + dd];
        }
        acc = acc * cst[576 + ch] + cst[640 + ch];
        h1s[pos * 64 + ch] = (_Float16)fmaxf(acc, 0.0f);
    }
    __syncthreads();

    // ---- Phase 2: conv2 (64->32) implicit-GEMM WMMA: 81 N-tiles, K = 9dd x 2 chunks ----
    for (int t = wid; t < 81; t += 8) {
        int p  = t * 16 + ln;
        int py = p / H2D, px = p % H2D;
        v8f acc0 = {0, 0, 0, 0, 0, 0, 0, 0};
        v8f acc1 = {0, 0, 0, 0, 0, 0, 0, 0};
#pragma unroll
        for (int dd = 0; dd < 9; ++dd) {
            int bbase = ((py + dd / 3) * H1D + (px + dd % 3)) * 64 + lh * 16;
#pragma unroll
            for (int cc = 0; cc < 2; ++cc) {
                v16h bf = *(const v16h*)(h1s + bbase + cc * 32);
                v16h a0 = *(const v16h*)(aswz + (((dd * 2 + cc) * 2 + 0) * 32 + lane) * 16);
                v16h a1 = *(const v16h*)(aswz + (((dd * 2 + cc) * 2 + 1) * 32 + lane) * 16);
                acc0 = wmma_f16(a0, bf, acc0);
                acc1 = wmma_f16(a1, bf, acc1);
            }
        }
        v8h o0, o1;
#pragma unroll
        for (int r = 0; r < 8; ++r) {
            int c0 = lh * 8 + r, c1 = 16 + lh * 8 + r;
            o0[r] = (_Float16)fmaxf(acc0[r] * cst[704 + c0] + cst[736 + c0], 0.0f);
            o1[r] = (_Float16)fmaxf(acc1[r] * cst[704 + c1] + cst[736 + c1], 0.0f);
        }
        *(v8h*)(h2s + p * 32 + lh * 8)      = o0;
        *(v8h*)(h2s + p * 32 + 16 + lh * 8) = o1;
    }
    __syncthreads();

    // ---- rebuild A fragments for conv3 (32->32): block = dd*2+mt, single 32-ch K chunk ----
    for (int i = tid; i < 18 * 512; i += 256) {
        int blk = i >> 9, l = (i >> 4) & 31, s = i & 15;
        int mt = blk & 1, dd = blk >> 1;
        int K  = a_k_of(l, s);
        int m  = mt * 16 + (l & 15);
        aswz[i] = (_Float16)w3[(m * 32 + K) * 9 + dd];
    }
    __syncthreads();

    // ---- Phase 3: conv3 WMMA: 1156 positions -> 73 N-tiles (tail clamped) ----
    for (int t = wid; t < 73; t += 8) {
        int p  = t * 16 + ln;
        int pc = p < H3D * H3D ? p : H3D * H3D - 1;
        int py = pc / H3D, px = pc % H3D;
        v8f acc0 = {0, 0, 0, 0, 0, 0, 0, 0};
        v8f acc1 = {0, 0, 0, 0, 0, 0, 0, 0};
#pragma unroll
        for (int dd = 0; dd < 9; ++dd) {
            v16h bf = *(const v16h*)(h2s + ((py + dd / 3) * H2D + (px + dd % 3)) * 32 + lh * 16);
            v16h a0 = *(const v16h*)(aswz + ((dd * 2 + 0) * 32 + lane) * 16);
            v16h a1 = *(const v16h*)(aswz + ((dd * 2 + 1) * 32 + lane) * 16);
            acc0 = wmma_f16(a0, bf, acc0);
            acc1 = wmma_f16(a1, bf, acc1);
        }
        if (p < H3D * H3D) {
            v8h o0, o1;
#pragma unroll
            for (int r = 0; r < 8; ++r) {
                int c0 = lh * 8 + r, c1 = 16 + lh * 8 + r;
                o0[r] = (_Float16)fmaxf(acc0[r] * cst[768 + c0] + cst[800 + c0], 0.0f);
                o1[r] = (_Float16)fmaxf(acc1[r] * cst[768 + c1] + cst[800 + c1], 0.0f);
            }
            *(v8h*)(h3s + p * 32 + lh * 8)      = o0;
            *(v8h*)(h3s + p * 32 + 16 + lh * 8) = o1;
        }
    }
    __syncthreads();

    // ---- Phase 4: conv4 (32->1) fp32 + sigmoid + per-cell stats ----
    const float b4v = cst[1120];
    for (int i = tid; i < TILE * TILE; i += 256) {
        int ry = i >> 5, rx = i & 31;
        float acc = b4v;
#pragma unroll
        for (int dd = 0; dd < 9; ++dd) {
            const _Float16* hp = h3s + ((ry + dd / 3) * H3D + (rx + dd % 3)) * 32;
#pragma unroll
            for (int c = 0; c < 32; ++c) acc += (float)hp[c] * cst[832 + c * 9 + dd];
        }
        int gy = gy0 + ry, gx = gx0 + rx;
        rawOut[(size_t)b * HW * HW + gy * HW + gx] = acc;
        float sg = 1.0f / (1.0f + __expf(-acc));
        int cell = (ry >> 3) * 4 + (rx >> 3);
        atomicMax((unsigned int*)&cst[1121 + cell], __float_as_uint(sg));
        atomicAdd(&cst[1137 + cell], sg);
        atomicAdd(&cst[1153 + cell], sg * ((float)gx * (1.0f / 127.0f)));
        atomicAdd(&cst[1169 + cell], sg * ((float)gy * (1.0f / 127.0f)));
    }
    __syncthreads();

    if (tid < 16) {
        int cy = tid >> 2, cx = tid & 3;
        int cellg = (ty * 4 + cy) * 16 + (tx * 4 + cx);
        float* st = stats + ((size_t)b * 256 + cellg) * 4;
        st[0] = cst[1121 + tid];
        st[1] = cst[1137 + tid];
        st[2] = cst[1153 + tid];
        st[3] = cst[1169 + tid];
    }
}

// ---- Stage 2: per-image top-17 threshold, valid mask, centroids, counts ----
__global__ __launch_bounds__(256)
void finalize_nodes(const float* __restrict__ stats,
                    float* __restrict__ nodes, float* __restrict__ valid,
                    float* __restrict__ npi) {
    __shared__ float g[256];
    __shared__ float red[256];
    int b = blockIdx.x, i = threadIdx.x;
    const float* st = stats + ((size_t)b * 256 + i) * 4;
    float mx = st[0], s = st[1], sx = st[2], sy = st[3];
    g[i] = mx;
    __syncthreads();

    int cge = 0;
    for (int j = 0; j < 256; ++j) cge += (g[j] >= mx) ? 1 : 0;
    // 17th largest = max value whose >=-count reaches 17
    red[i] = (cge >= 17) ? mx : -1e30f;
    __syncthreads();
    for (int stp = 128; stp > 0; stp >>= 1) {
        if (i < stp) red[i] = fmaxf(red[i], red[i + stp]);
        __syncthreads();
    }
    float thresh = fmaxf(red[0], 0.9f);
    int v = (mx > thresh) ? 1 : 0;

    nodes[((size_t)b * 256 + i) * 2 + 0] = sx / s;
    nodes[((size_t)b * 256 + i) * 2 + 1] = sy / s;
    valid[(size_t)b * 256 + i] = (float)v;

    __syncthreads();
    red[i] = (float)v;
    __syncthreads();
    for (int stp = 128; stp > 0; stp >>= 1) {
        if (i < stp) red[i] += red[i + stp];
        __syncthreads();
    }
    if (i == 0) npi[b] = red[0];
}

extern "C" void kernel_launch(void* const* d_in, const int* in_sizes, int n_in,
                              void* d_out, int out_size, void* d_ws, size_t ws_size,
                              hipStream_t stream) {
    const float* x   = (const float*)d_in[0];
    const float* w1  = (const float*)d_in[1];
    const float* b1  = (const float*)d_in[2];
    const float* g1  = (const float*)d_in[3];
    const float* bt1 = (const float*)d_in[4];
    const float* rm1 = (const float*)d_in[5];
    const float* rv1 = (const float*)d_in[6];
    const float* w2  = (const float*)d_in[7];
    const float* b2  = (const float*)d_in[8];
    const float* g2  = (const float*)d_in[9];
    const float* bt2 = (const float*)d_in[10];
    const float* rm2 = (const float*)d_in[11];
    const float* rv2 = (const float*)d_in[12];
    const float* w3  = (const float*)d_in[13];
    const float* b3  = (const float*)d_in[14];
    const float* g3  = (const float*)d_in[15];
    const float* bt3 = (const float*)d_in[16];
    const float* rm3 = (const float*)d_in[17];
    const float* rv3 = (const float*)d_in[18];
    const float* w4  = (const float*)d_in[19];
    const float* b4  = (const float*)d_in[20];

    float* out   = (float*)d_out;
    float* stats = (float*)d_ws;  // [128][256][4] fp32 = 512 KB

    fused_conv_tile<<<dim3(128 * 16), dim3(256), SMEM_BYTES, stream>>>(
        x, w1, b1, g1, bt1, rm1, rv1,
        w2, b2, g2, bt2, rm2, rv2,
        w3, b3, g3, bt3, rm3, rv3,
        w4, b4,
        out + OUT_RAW, stats);

    finalize_nodes<<<dim3(128), dim3(256), 0, stream>>>(
        stats, out + OUT_NODES, out + OUT_VALID, out + OUT_NPI);
}
